// SparseVarTransformer_38594576122321
// MI455X (gfx1250) — matrix operand order repaired and checked by reading, not compiled, not run on hardware
//
#include <hip/hip_runtime.h>
#include <hip/hip_bf16.h>
#include <stdint.h>

// ---------------- problem constants ----------------
constexpr int Bb  = 4;
constexpr int Ss  = 4096;
constexpr int DMm = 512;
constexpr int Hh  = 8;
constexpr int NBn = 16;
constexpr int BSs = 256;
constexpr int DHd = 64;
constexpr int KSk = 3;
constexpr int Mtot = Bb * Ss;                 // 16384
constexpr size_t NQKV = (size_t)Mtot * DMm;   // 8388608 elements

// ---------------- vector types ----------------
typedef __attribute__((ext_vector_type(16))) __bf16 v16bf;
typedef __attribute__((ext_vector_type(8)))  __bf16 v8bf;
typedef __attribute__((ext_vector_type(8)))  float  v8f;
typedef uint32_t u32x4 __attribute__((ext_vector_type(4)));
typedef int      i32x8 __attribute__((ext_vector_type(8)));
typedef int      i32x4 __attribute__((ext_vector_type(4)));

// ---------------- gfx1250 async / TDM feature detection ----------------
#if defined(__HIP_DEVICE_COMPILE__) && __has_builtin(__builtin_amdgcn_global_load_async_to_lds_b128)
#define HAVE_ASYNC_LDS 1
#endif
#if defined(__HIP_DEVICE_COMPILE__) && __has_builtin(__builtin_amdgcn_tensor_load_to_lds) && \
    __has_builtin(__builtin_amdgcn_s_wait_tensorcnt)
#define HAVE_TDM 1
#endif

// payload-typed, addrspace-qualified pointers for the async builtins
typedef __attribute__((address_space(1))) i32x4 g_v4i;   // global
typedef __attribute__((address_space(3))) i32x4 l_v4i;   // LDS
typedef __attribute__((address_space(3))) const char lds_cchar_t;

__device__ __forceinline__ uint32_t lds_addr_of(const void* p) {
  return (uint32_t)(size_t)(lds_cchar_t*)p;
}

__device__ __forceinline__ void async_copy_b128(void* lds_dst, const void* gsrc) {
#ifdef HAVE_ASYNC_LDS
  __builtin_amdgcn_global_load_async_to_lds_b128((g_v4i*)gsrc, (l_v4i*)lds_dst, 0, 0);
#else
  (void)lds_dst; (void)gsrc;
#endif
}

__device__ __forceinline__ void wait_async0() {
#if defined(__HIP_DEVICE_COMPILE__) && __has_builtin(__builtin_amdgcn_s_wait_asynccnt)
  __builtin_amdgcn_s_wait_asynccnt(0);
#elif defined(__HIP_DEVICE_COMPILE__)
  asm volatile("s_wait_asynccnt 0" ::: "memory");
#endif
}

__device__ __forceinline__ v8f zero_v8f() {
  v8f z;
#pragma unroll
  for (int i = 0; i < 8; ++i) z[i] = 0.0f;
  return z;
}

__device__ __forceinline__ v8bf zero_v8bf() {
  v8bf z;
#pragma unroll
  for (int i = 0; i < 8; ++i) z[i] = (__bf16)0.0f;
  return z;
}

// A fragment (16x32 bf16): lane&15 = row M; lane>>4 selects K halves.
__device__ __forceinline__ v16bf load_a_frag(const __bf16* base, int ld) {
  int lane = threadIdx.x & 31;
  int row = lane & 15, kg = lane >> 4;
  v8bf lo = *(const v8bf*)(base + row * ld + kg * 8);
  v8bf hi = *(const v8bf*)(base + row * ld + kg * 8 + 16);
  return __builtin_shufflevector(lo, hi, 0,1,2,3,4,5,6,7,8,9,10,11,12,13,14,15);
}

// B fragment (32x16 bf16), sourced from B^T layout [N][K] (K contiguous).
__device__ __forceinline__ v16bf load_b_frag(const __bf16* baseT, int ld) {
  int lane = threadIdx.x & 31;
  int col = lane & 15, kg = lane >> 4;
  const __bf16* p = baseT + col * ld + kg * 8;
  v8bf lo = *(const v8bf*)(p);
  v8bf hi = *(const v8bf*)(p + 16);
  return __builtin_shufflevector(lo, hi, 0,1,2,3,4,5,6,7,8,9,10,11,12,13,14,15);
}

__device__ __forceinline__ v8f bf16_wmma(v16bf a, v16bf b, v8f c) {
  return __builtin_amdgcn_wmma_f32_16x16x32_bf16(false, a, false, b, (short)0, c,
                                                 false, false);
}

// ---------------- kernel 1: fp32 -> bf16 conversion + mask prep ----------------
__global__ void prep_kernel(const float* __restrict__ q, const float* __restrict__ k,
                            const float* __restrict__ v, const float* __restrict__ mask,
                            const float* __restrict__ Wq, const float* __restrict__ Wk,
                            const float* __restrict__ Wv, const float* __restrict__ Wo,
                            __bf16* __restrict__ qb, __bf16* __restrict__ kb,
                            __bf16* __restrict__ vb, __bf16* __restrict__ wqb,
                            __bf16* __restrict__ wkb, __bf16* __restrict__ wvb,
                            __bf16* __restrict__ wob, float* __restrict__ u) {
  size_t i0 = (size_t)blockIdx.x * blockDim.x + threadIdx.x;
  size_t stride = (size_t)gridDim.x * blockDim.x;
  for (size_t i = i0; i < NQKV; i += stride) {
    qb[i] = (__bf16)q[i];
    kb[i] = (__bf16)k[i];
    vb[i] = (__bf16)v[i];
    if (i < (size_t)KSk * DMm * DMm) wqb[i] = (__bf16)Wq[i];
    if (i < (size_t)DMm * DMm) {
      wkb[i] = (__bf16)Wk[i];
      wvb[i] = (__bf16)Wv[i];
      wob[i] = (__bf16)Wo[i];
    }
    if (i < (size_t)Bb * Ss) u[i] = __expf(-1e9f * mask[i]);
  }
}

// ---------------- kernel 2: 128x128-tile WMMA GEMM ----------------
template <int TAPS, bool F32OUT>
__global__ __launch_bounds__(256) void gemm_kernel(const __bf16* __restrict__ X,
                                                   const __bf16* __restrict__ W,
                                                   const float* __restrict__ bias,
                                                   __bf16* __restrict__ Ybf,
                                                   float* __restrict__ Yf) {
  constexpr int LD = 40;            // padded LDS stride (halves)
  __shared__ __bf16 As[128 * LD];
  __shared__ __bf16 Bs[128 * LD];   // stored transposed: Bs[n][k]

  const int m0 = blockIdx.x * 128;
  const int n0 = blockIdx.y * 128;
  const int tid = threadIdx.x;
  const int wave = tid >> 5, lane = tid & 31;
  const int wm = wave >> 1, wn = wave & 1;
  const int hi = lane >> 4, l15 = lane & 15;

  v8f acc[2][4];
#pragma unroll
  for (int mi = 0; mi < 2; ++mi)
#pragma unroll
    for (int ni = 0; ni < 4; ++ni) acc[mi][ni] = zero_v8f();

#pragma unroll 1
  for (int tap = 0; tap < TAPS; ++tap) {
    const __bf16* Wt = W + (size_t)tap * DMm * DMm;
    const int shift = tap - (TAPS - 1);  // <= 0, rows before batch start are zero
#pragma unroll 1
    for (int k0 = 0; k0 < DMm; k0 += 32) {
      // stage A tile: 128 rows x 32 (bf16) -- async DMA into LDS when available
#pragma unroll
      for (int i = 0; i < 2; ++i) {
        int idx = tid + i * 256;            // 0..511
        int r = idx >> 2, c8 = (idx & 3) * 8;
        int m = m0 + r;
        int s_in_batch = (m & (Ss - 1)) + shift;
#ifdef HAVE_ASYNC_LDS
        if (s_in_batch >= 0) {
          async_copy_b128(As + r * LD + c8,
                          X + (size_t)(m + shift) * DMm + k0 + c8);
        } else {
          *(v8bf*)(As + r * LD + c8) = zero_v8bf();
        }
#else
        v8bf av = zero_v8bf();
        if (s_in_batch >= 0)
          av = *(const v8bf*)(X + (size_t)(m + shift) * DMm + k0 + c8);
        *(v8bf*)(As + r * LD + c8) = av;
#endif
      }
      // stage B tile transposed: Bs[n][k]
#pragma unroll
      for (int i = 0; i < 2; ++i) {
        int idx = tid + i * 256;            // 0..511
        int kk = idx >> 4, n8 = (idx & 15) * 8;
        v8bf bvv = *(const v8bf*)(Wt + (size_t)(k0 + kk) * DMm + n0 + n8);
#pragma unroll
        for (int j = 0; j < 8; ++j) Bs[(n8 + j) * LD + kk] = bvv[j];
      }
      // prefetch next A stripe (global_prefetch_b8)
      if (k0 + 32 < DMm)
        __builtin_prefetch(X + (size_t)(m0 + (tid >> 1)) * DMm + k0 + 32, 0, 1);
      wait_async0();
      __syncthreads();

#pragma unroll
      for (int mi = 0; mi < 2; ++mi) {
        v16bf a = load_a_frag(As + (wm * 32 + mi * 16) * LD, LD);
#pragma unroll
        for (int ni = 0; ni < 4; ++ni) {
          v16bf b = load_b_frag(Bs + (wn * 64 + ni * 16) * LD, LD);
          acc[mi][ni] = bf16_wmma(a, b, acc[mi][ni]);
        }
      }
      __syncthreads();
    }
  }

  // epilogue
#pragma unroll
  for (int mi = 0; mi < 2; ++mi)
#pragma unroll
    for (int ni = 0; ni < 4; ++ni)
#pragma unroll
      for (int j = 0; j < 8; ++j) {
        int m = m0 + wm * 32 + mi * 16 + j + 8 * hi;
        int nidx = n0 + wn * 64 + ni * 16 + l15;
        float val = acc[mi][ni][j] + bias[nidx];
        if constexpr (F32OUT) {
          Yf[(size_t)m * DMm + nidx] = val;
        } else {
          int bb = m >> 12, s = m & (Ss - 1);
          int hh = nidx >> 6, d = nidx & 63;
          Ybf[(((size_t)(bb * Hh + hh)) * Ss + s) * DHd + d] = (__bf16)val;
        }
      }
}

// ---------------- kernel 3: per-block v sums ----------------
__global__ void reduce_blocks(const __bf16* __restrict__ vh, const float* __restrict__ u,
                              float* __restrict__ uvb, float* __restrict__ cnt_blk) {
  int bhn = blockIdx.x;              // [B*H*NB]
  int d = threadIdx.x;               // 0..63
  int n = bhn & 15, h = (bhn >> 4) & 7, b = bhn >> 7;
  const __bf16* vp = vh + (((size_t)(b * Hh + h)) * Ss + n * BSs) * DHd;
  const float* up = u + (size_t)b * Ss + n * BSs;
  float s = 0.0f;
  for (int kk = 0; kk < BSs; ++kk) s += up[kk] * (float)vp[(size_t)kk * DHd + d];
  uvb[(size_t)bhn * DHd + d] = s;
  if (h == 0 && d == 0) {
    float c = 0.0f;
    for (int kk = 0; kk < BSs; ++kk) c += up[kk];
    cnt_blk[b * NBn + n] = c;
  }
}

__global__ void reduce_all(const float* __restrict__ uvb, const float* __restrict__ cnt_blk,
                           float* __restrict__ uv_all, float* __restrict__ cnt_all) {
  int bh = blockIdx.x;               // [B*H]
  int d = threadIdx.x;               // 0..63
  float s = 0.0f;
  for (int n = 0; n < NBn; ++n) s += uvb[((size_t)(bh * NBn + n)) * DHd + d];
  uv_all[(size_t)bh * DHd + d] = s;
  if (d == 0 && (bh & 7) == 0) {
    int b = bh >> 3;
    float c = 0.0f;
    for (int n = 0; n < NBn; ++n) c += cnt_blk[b * NBn + n];
    cnt_all[b] = c;
  }
}

// ---------------- kernel 4: fused block attention ----------------
constexpr int VT_LD = 264;  // padded stride for V^T in LDS (halves)
constexpr size_t ATT_SMEM =
    (size_t)(BSs * DHd + DHd * VT_LD + 8 * 16 * BSs) * sizeof(__bf16) + BSs * sizeof(float);

__global__ __launch_bounds__(256) void attn_kernel(
    const __bf16* __restrict__ qh, const __bf16* __restrict__ kh,
    const __bf16* __restrict__ vh, const float* __restrict__ mask,
    const float* __restrict__ uvb, const float* __restrict__ uv_all,
    const float* __restrict__ cnt_blk, const float* __restrict__ cnt_all,
    __bf16* __restrict__ A2) {
  extern __shared__ __align__(16) char smem_raw[];
  __bf16* k_lds = (__bf16*)smem_raw;           // [256][64]
  __bf16* vT    = k_lds + BSs * DHd;           // [64][VT_LD]
  __bf16* e_lds = vT + DHd * VT_LD;            // [8 waves][16][256]
  float*  ma    = (float*)(e_lds + 8 * 16 * BSs);  // [256]

  const int bhn = blockIdx.x;
  const int n = bhn & 15, h = (bhn >> 4) & 7, b = bhn >> 7;
  const size_t head_off = (((size_t)(b * Hh + h)) * Ss + (size_t)n * BSs) * DHd;
  const __bf16* qptr = qh + head_off;
  const __bf16* kptr = kh + head_off;
  const __bf16* vptr = vh + head_off;
  const int tid = threadIdx.x;

  // ---- stage K block [256][64] (contiguous 32KB tile) ----
#ifdef HAVE_TDM
  if (tid == 0) {
    // D# group 0: count=1, lds_addr, 57-bit global addr, type=2
    uint64_t ga = (uint64_t)(size_t)kptr;
    u32x4 g0;
    g0[0] = 1u;                               // count=1, user mode
    g0[1] = lds_addr_of(k_lds);               // LDS byte address
    g0[2] = (uint32_t)ga;                     // global addr [31:0]
    g0[3] = (uint32_t)((ga >> 32) & 0x01FFFFFFu) | (2u << 30);  // [56:32] | type=2
    // D# group 1: data_size=2B; tensor_dim0=tile_dim0=16384; tensor_dim1=1;
    // tile_dim1=1; tensor_dim0_stride=16384  (bit-packed per ISA p.153)
    i32x8 g1;
    g1[0] = 0x00010000;   // workgroup_mask=0 | data_size=1 (2 bytes) << 16
    g1[1] = 0x40000000;   // tensor_dim0[15:0]=0x4000 at bits 63:48
    g1[2] = 0x00010000;   // tensor_dim0[31:16]=0 | tensor_dim1[15:0]=1 at bits 95:80
    g1[3] = 0x40000000;   // tensor_dim1[31:16]=0 | tile_dim0=0x4000 at bits 127:112
    g1[4] = 0x00000001;   // tile_dim1=1 | tile_dim2=0
    g1[5] = 16384;        // tensor_dim0_stride[31:0]
    g1[6] = 0;            // tensor_dim0_stride[47:32] | tensor_dim1_stride[15:0]
    g1[7] = 0;            // tensor_dim1_stride[47:16]
    i32x4 g2 = {0, 0, 0, 0};
    i32x4 g3 = {0, 0, 0, 0};
#if __clang_major__ >= 23
    i32x8 g4 = {0, 0, 0, 0, 0, 0, 0, 0};
    __builtin_amdgcn_tensor_load_to_lds(g0, g1, g2, g3, g4, 0);
#else
    __builtin_amdgcn_tensor_load_to_lds(g0, g1, g2, g3, 0);
#endif
    __builtin_amdgcn_s_wait_tensorcnt(0);
  }
#elif defined(HAVE_ASYNC_LDS)
#pragma unroll
  for (int i = 0; i < 8; ++i) {
    int idx = tid + i * 256;                   // 2048 x 16B
    async_copy_b128((__bf16*)k_lds + idx * 8, kptr + (size_t)idx * 8);
  }
#else
#pragma unroll
  for (int i = 0; i < 8; ++i) {
    int idx = tid + i * 256;
    ((v8bf*)k_lds)[idx] = ((const v8bf*)kptr)[idx];
  }
#endif

  // ---- stage V^T [64][256] (transpose; register round-trip) ----
#pragma unroll
  for (int i = 0; i < 8; ++i) {
    int idx = tid + i * 256;
    int kk = idx >> 3, d8 = (idx & 7) * 8;
    v8bf vv = *(const v8bf*)(vptr + (size_t)kk * DHd + d8);
#pragma unroll
    for (int j = 0; j < 8; ++j) vT[(d8 + j) * VT_LD + kk] = vv[j];
  }
  ma[tid] = -1e9f * mask[(size_t)b * Ss + n * BSs + tid];
#if !defined(HAVE_TDM) && defined(HAVE_ASYNC_LDS)
  wait_async0();
#endif
  __syncthreads();

  const float cnt_off = cnt_all[b] - cnt_blk[b * NBn + n];
  const int wave = tid >> 5, lane = tid & 31;
  const int hi = lane >> 4, l15 = lane & 15;
  __bf16* ew = e_lds + wave * (16 * BSs);

#pragma unroll 1
  for (int pass = 0; pass < 2; ++pass) {
    const int r0 = pass * 128 + wave * 16;     // 16 q-rows per wave per pass
    const __bf16* qrow = qptr + (size_t)r0 * DHd;
    v16bf aq0 = load_a_frag(qrow, DHd);        // dh 0..31
    v16bf aq1 = load_a_frag(qrow + 32, DHd);   // dh 32..63

    // QK^T: 16 key tiles of 16
    v8f acc[16];
#pragma unroll
    for (int t = 0; t < 16; ++t) {
      v8f c = zero_v8f();
      c = bf16_wmma(aq0, load_b_frag(k_lds + t * 16 * DHd, DHd), c);
      c = bf16_wmma(aq1, load_b_frag(k_lds + t * 16 * DHd + 32, DHd), c);
      acc[t] = c;
    }
    // scale + additive key mask
#pragma unroll
    for (int t = 0; t < 16; ++t) {
      float madd = ma[t * 16 + l15];
#pragma unroll
      for (int j = 0; j < 8; ++j) acc[t][j] = acc[t][j] * 0.125f + madd;
    }
    // row max (include the implicit off-block logit 0)
    float rowM[8], dn[8], eM[8];
#pragma unroll
    for (int j = 0; j < 8; ++j) {
      float r = acc[0][j];
#pragma unroll
      for (int t = 1; t < 16; ++t) r = fmaxf(r, acc[t][j]);
      r = fmaxf(r, __shfl_xor(r, 1, 32));
      r = fmaxf(r, __shfl_xor(r, 2, 32));
      r = fmaxf(r, __shfl_xor(r, 4, 32));
      r = fmaxf(r, __shfl_xor(r, 8, 32));
      rowM[j] = fmaxf(r, 0.0f);
      eM[j] = __expf(-rowM[j]);
    }
    // exponentiate + denom
#pragma unroll
    for (int j = 0; j < 8; ++j) {
      float s = 0.0f;
#pragma unroll
      for (int t = 0; t < 16; ++t) {
        float e = __expf(acc[t][j] - rowM[j]);
        acc[t][j] = e;
        s += e;
      }
      s += __shfl_xor(s, 1, 32);
      s += __shfl_xor(s, 2, 32);
      s += __shfl_xor(s, 4, 32);
      s += __shfl_xor(s, 8, 32);
      dn[j] = s + cnt_off * eM[j];
    }
    // spill e (bf16) to LDS in A-source layout [row][key]
#pragma unroll
    for (int t = 0; t < 16; ++t)
#pragma unroll
      for (int j = 0; j < 8; ++j)
        ew[(j + 8 * hi) * BSs + t * 16 + l15] = (__bf16)acc[t][j];
    __syncthreads();

    // PV: out tile 16 x 64
#pragma unroll
    for (int nt = 0; nt < 4; ++nt) {
      v8f o = zero_v8f();
#pragma unroll
      for (int kt = 0; kt < 8; ++kt) {
        v16bf a = load_a_frag(ew + kt * 32, BSs);
        v16bf bfr = load_b_frag(vT + nt * 16 * VT_LD + kt * 32, VT_LD);
        o = bf16_wmma(a, bfr, o);
      }
      int d = nt * 16 + l15;
      float ua = uv_all[(size_t)(b * Hh + h) * DHd + d];
      float ub = uvb[(size_t)bhn * DHd + d];
#pragma unroll
      for (int j = 0; j < 8; ++j) {
        float numv = o[j] + eM[j] * (ua - ub);
        float attn = numv / dn[j];
        int srow = n * BSs + r0 + j + 8 * hi;
        A2[((size_t)b * Ss + srow) * DMm + h * DHd + d] = (__bf16)attn;
      }
    }
    __syncthreads();  // before e_lds is overwritten next pass
  }
}

// ---------------- host-side launch ----------------
extern "C" void kernel_launch(void* const* d_in, const int* in_sizes, int n_in,
                              void* d_out, int out_size, void* d_ws, size_t ws_size,
                              hipStream_t stream) {
  (void)in_sizes; (void)n_in; (void)out_size; (void)ws_size;
  const float* q    = (const float*)d_in[0];
  const float* k    = (const float*)d_in[1];
  const float* v    = (const float*)d_in[2];
  const float* mask = (const float*)d_in[3];
  const float* Wq   = (const float*)d_in[4];
  const float* bq   = (const float*)d_in[5];
  const float* Wk   = (const float*)d_in[6];
  const float* bk   = (const float*)d_in[7];
  const float* Wv   = (const float*)d_in[8];
  const float* bv   = (const float*)d_in[9];
  const float* Wo   = (const float*)d_in[10];
  const float* bo   = (const float*)d_in[11];
  float* out = (float*)d_out;

  char* ws = (char*)d_ws;
  size_t off = 0;
  auto alloc = [&](size_t bytes) {
    char* p = ws + off;
    off += (bytes + 255) & ~(size_t)255;
    return p;
  };
  __bf16* qb  = (__bf16*)alloc(NQKV * 2);
  __bf16* kb  = (__bf16*)alloc(NQKV * 2);
  __bf16* vb  = (__bf16*)alloc(NQKV * 2);
  __bf16* wqb = (__bf16*)alloc((size_t)KSk * DMm * DMm * 2);
  __bf16* wkb = (__bf16*)alloc((size_t)DMm * DMm * 2);
  __bf16* wvb = (__bf16*)alloc((size_t)DMm * DMm * 2);
  __bf16* wob = (__bf16*)alloc((size_t)DMm * DMm * 2);
  __bf16* qhp = (__bf16*)alloc(NQKV * 2);
  __bf16* khp = (__bf16*)alloc(NQKV * 2);
  __bf16* vhp = (__bf16*)alloc(NQKV * 2);
  __bf16* A2  = (__bf16*)alloc(NQKV * 2);
  float* u       = (float*)alloc((size_t)Bb * Ss * 4);
  float* uvb     = (float*)alloc((size_t)Bb * Hh * NBn * DHd * 4);
  float* uv_all  = (float*)alloc((size_t)Bb * Hh * DHd * 4);
  float* cnt_blk = (float*)alloc((size_t)Bb * NBn * 4);
  float* cnt_all = (float*)alloc((size_t)Bb * 4);

  prep_kernel<<<2048, 256, 0, stream>>>(q, k, v, mask, Wq, Wk, Wv, Wo,
                                        qb, kb, vb, wqb, wkb, wvb, wob, u);

  dim3 ggrid(Mtot / 128, DMm / 128);
  gemm_kernel<1, false><<<ggrid, 256, 0, stream>>>(kb, wkb, bk, khp, nullptr);
  gemm_kernel<1, false><<<ggrid, 256, 0, stream>>>(vb, wvb, bv, vhp, nullptr);
  gemm_kernel<3, false><<<ggrid, 256, 0, stream>>>(qb, wqb, bq, qhp, nullptr);

  reduce_blocks<<<Bb * Hh * NBn, DHd, 0, stream>>>(vhp, u, uvb, cnt_blk);
  reduce_all<<<Bb * Hh, DHd, 0, stream>>>(uvb, cnt_blk, uv_all, cnt_all);

  attn_kernel<<<Bb * Hh * NBn, 256, ATT_SMEM, stream>>>(qhp, khp, vhp, mask, uvb,
                                                        uv_all, cnt_blk, cnt_all, A2);

  gemm_kernel<1, true><<<ggrid, 256, 0, stream>>>(A2, wob, bo, nullptr, out);
}
